// FusedSigmaMLP_57054345560720
// MI455X (gfx1250) — compile-verified, matrix operand order checked by reading, not run on hardware
//
#include <hip/hip_runtime.h>
#include <stdint.h>

// Problem constants (from reference): E=8, T=1024, H=2048, I=1408
#define E_ 8
#define T_ 1024
#define H_ 2048
#define I_ 1408

// Trivial vector types (safe inside unions, unlike HIP's struct float4)
typedef __attribute__((ext_vector_type(4)))  float        f4;
typedef __attribute__((ext_vector_type(4)))  unsigned int u4;
typedef __attribute__((ext_vector_type(4)))  int          i4;
typedef __attribute__((ext_vector_type(8)))  float        v8f;
typedef __attribute__((ext_vector_type(2)))  __bf16       v2bf;
typedef __attribute__((ext_vector_type(16))) __bf16       v16bf;

union Frag16 {
    u4    u[2];   // two ds_load_b128 worth of data
    v16bf v;      // WMMA 16-bit A/B operand (8 VGPRs)
};

// gfx1250 async copy global->LDS (ASYNCcnt path), guarded so we always compile
#if __has_builtin(__builtin_amdgcn_global_load_async_to_lds_b128)
#define HAVE_ASYNC_LDS 1
typedef __attribute__((address_space(1))) i4* g4p_t;  // global v4i*
typedef __attribute__((address_space(3))) i4* l4p_t;  // LDS v4i*
__device__ __forceinline__ void wait_asynccnt0() {
#if __has_builtin(__builtin_amdgcn_s_wait_asynccnt)
    __builtin_amdgcn_s_wait_asynccnt(0);
#else
    asm volatile("s_wait_asynccnt 0x0" ::: "memory");
#endif
}
#endif

// pack two f32 (k, k+1) into one dword of bf16 (lo = even k).
// Native __bf16 fptrunc lets the backend use v_cvt_pk_bf16_f32-class ops.
__device__ __forceinline__ unsigned pack2(float lo, float hi) {
    v2bf v;
    v.x = (__bf16)lo;
    v.y = (__bf16)hi;
    return __builtin_bit_cast(unsigned, v);
}
__device__ __forceinline__ unsigned short f2bf(float f) {
    __bf16 b = (__bf16)f;
    return __builtin_bit_cast(unsigned short, b);
}

// ---------------------------------------------------------------------------
// Kernel 1: act[e,t,i] = bf16( silu(x@gate) * (x@up) )
// Block tile: 128(T) x 128(I), K-step 32 over H. 8 waves, each 32x64,
// A tile staged once and reused for BOTH weight matrices.
// ---------------------------------------------------------------------------
__global__ __launch_bounds__(256)
void gateup_kernel(const float* __restrict__ x,
                   const float* __restrict__ gate,
                   const float* __restrict__ up,
                   unsigned short* __restrict__ act) {
    __shared__ __align__(16) unsigned short sA[128 * 32]; // [m][k] bf16
    __shared__ __align__(16) unsigned short sG[128 * 32]; // [n][k] bf16 (transposed)
    __shared__ __align__(16) unsigned short sU[128 * 32]; // [n][k] bf16 (transposed)

    const int e    = blockIdx.z;
    const int tm0  = blockIdx.y * 128;
    const int n0   = blockIdx.x * 128;
    const int tid  = threadIdx.x;
    const int lane = tid & 31;
    const int wave = tid >> 5;
    const int wm   = wave & 3;   // m offset = 32*wm
    const int wn   = wave >> 2;  // n offset = 64*wn
    const int lrow = lane & 15;
    const int khalf = lane >> 4; // 0: K0-7/16-23, 1: K8-15/24-31

    const float* X = x    + ((size_t)e * T_ + tm0) * H_;
    const float* G = gate + (size_t)e * H_ * I_;
    const float* U = up   + (size_t)e * H_ * I_;

    v8f accG[2][4], accU[2][4];
#pragma unroll
    for (int mi = 0; mi < 2; ++mi)
#pragma unroll
        for (int ni = 0; ni < 4; ++ni) {
            v8f z = {0.f, 0.f, 0.f, 0.f, 0.f, 0.f, 0.f, 0.f};
            accG[mi][ni] = z;
            accU[mi][ni] = z;
        }

    const int arow  = tid >> 1; // 0..127
    const int ahalf = tid & 1;  // which 16-wide K segment

    for (int k0 = 0; k0 < H_; k0 += 32) {
        __syncthreads();

        // ---- stage A tile: x[tm0..+128, k0..+32] -> sA bf16 [m][k]
        {
            const float* ap = X + (size_t)arow * H_ + k0 + ahalf * 16;
            f4 f0 = ((const f4*)ap)[0];
            f4 f1 = ((const f4*)ap)[1];
            f4 f2 = ((const f4*)ap)[2];
            f4 f3 = ((const f4*)ap)[3];
            u4 p0, p1;
            p0.x = pack2(f0.x, f0.y); p0.y = pack2(f0.z, f0.w);
            p0.z = pack2(f1.x, f1.y); p0.w = pack2(f1.z, f1.w);
            p1.x = pack2(f2.x, f2.y); p1.y = pack2(f2.z, f2.w);
            p1.z = pack2(f3.x, f3.y); p1.w = pack2(f3.z, f3.w);
            *(u4*)&sA[arow * 32 + ahalf * 16]     = p0;
            *(u4*)&sA[arow * 32 + ahalf * 16 + 8] = p1;
            if (k0 + 32 < H_) __builtin_prefetch(ap + 32, 0, 0); // global_prefetch_b8
        }

        // ---- stage gate/up tiles transposed: W[k0..+32, n0..+128] -> s[n][k]
#pragma unroll
        for (int r = 0; r < 2; ++r) {
            int t  = tid + (r << 8);     // 0..511 tasks
            int kp = t & 15;             // k pair index (k = 2*kp)
            int ng = t >> 4;             // n group of 4 (0..31)
            const float* gp = G + (size_t)(k0 + kp * 2) * I_ + n0 + ng * 4;
            const float* pp = U + (size_t)(k0 + kp * 2) * I_ + n0 + ng * 4;
            f4 g0 = ((const f4*)gp)[0];
            f4 g1 = *(const f4*)(gp + I_);
            f4 u0 = ((const f4*)pp)[0];
            f4 u1 = *(const f4*)(pp + I_);
            int base = (ng * 4) * 32 + kp * 2; // halfword index, dword aligned
            *(unsigned*)&sG[base     ] = pack2(g0.x, g1.x);
            *(unsigned*)&sG[base + 32] = pack2(g0.y, g1.y);
            *(unsigned*)&sG[base + 64] = pack2(g0.z, g1.z);
            *(unsigned*)&sG[base + 96] = pack2(g0.w, g1.w);
            *(unsigned*)&sU[base     ] = pack2(u0.x, u1.x);
            *(unsigned*)&sU[base + 32] = pack2(u0.y, u1.y);
            *(unsigned*)&sU[base + 64] = pack2(u0.z, u1.z);
            *(unsigned*)&sU[base + 96] = pack2(u0.w, u1.w);
            if (k0 + 32 < H_) {
                __builtin_prefetch(gp + (size_t)32 * I_, 0, 0);
                __builtin_prefetch(pp + (size_t)32 * I_, 0, 0);
            }
        }
        __syncthreads();

        // ---- compute: 2 A frags, 4x(G,U) B frags, 16 WMMA per wave per k-step
        Frag16 a[2];
#pragma unroll
        for (int mi = 0; mi < 2; ++mi) {
            int base = (wm * 32 + mi * 16 + lrow) * 32 + khalf * 8;
            a[mi].u[0] = *(const u4*)&sA[base];
            a[mi].u[1] = *(const u4*)&sA[base + 16];
        }
#pragma unroll
        for (int ni = 0; ni < 4; ++ni) {
            int nb = (wn * 64 + ni * 16 + lrow) * 32 + khalf * 8;
            Frag16 bg, bu;
            bg.u[0] = *(const u4*)&sG[nb];
            bg.u[1] = *(const u4*)&sG[nb + 16];
            bu.u[0] = *(const u4*)&sU[nb];
            bu.u[1] = *(const u4*)&sU[nb + 16];
#pragma unroll
            for (int mi = 0; mi < 2; ++mi) {
                accG[mi][ni] = __builtin_amdgcn_wmma_f32_16x16x32_bf16(
                    false, a[mi].v, false, bg.v, (short)0, accG[mi][ni], false, false);
                accU[mi][ni] = __builtin_amdgcn_wmma_f32_16x16x32_bf16(
                    false, a[mi].v, false, bu.v, (short)0, accU[mi][ni], false, false);
            }
        }
    }

    // ---- epilogue: silu(g)*u, bf16 store to workspace activations
#pragma unroll
    for (int mi = 0; mi < 2; ++mi) {
#pragma unroll
        for (int ni = 0; ni < 4; ++ni) {
            int nn = n0 + wn * 64 + ni * 16 + lrow;
#pragma unroll
            for (int r = 0; r < 8; ++r) {
                float g = accG[mi][ni][r];
                float u = accU[mi][ni][r];
                float s = g * __builtin_amdgcn_rcpf(1.0f + __expf(-g)) * u;
                int m = tm0 + wm * 32 + mi * 16 + khalf * 8 + r; // C layout: lanes16-31 hold M+8
                act[((size_t)e * T_ + m) * I_ + nn] = f2bf(s);
            }
        }
    }
}

// ---------------------------------------------------------------------------
// Kernel 2: out[e,t,h] = act[e] (bf16, TxI) @ down_proj[e] (IxH)
// A tile is a pure bf16 copy -> use gfx1250 async global->LDS path.
// ---------------------------------------------------------------------------
__global__ __launch_bounds__(256)
void down_kernel(const unsigned short* __restrict__ act,
                 const float* __restrict__ down,
                 float* __restrict__ out) {
    __shared__ __align__(16) unsigned short sA[128 * 32]; // [m][k] bf16
    __shared__ __align__(16) unsigned short sB[128 * 32]; // [n][k] bf16 (transposed)

    const int e    = blockIdx.z;
    const int tm0  = blockIdx.y * 128;
    const int n0   = blockIdx.x * 128; // over H
    const int tid  = threadIdx.x;
    const int lane = tid & 31;
    const int wave = tid >> 5;
    const int wm   = wave & 3;
    const int wn   = wave >> 2;
    const int lrow = lane & 15;
    const int khalf = lane >> 4;

    const unsigned short* A = act + ((size_t)e * T_ + tm0) * I_;
    const float* D = down + (size_t)e * I_ * H_;

    v8f acc[2][4];
#pragma unroll
    for (int mi = 0; mi < 2; ++mi)
#pragma unroll
        for (int ni = 0; ni < 4; ++ni) {
            v8f z = {0.f, 0.f, 0.f, 0.f, 0.f, 0.f, 0.f, 0.f};
            acc[mi][ni] = z;
        }

    for (int k0 = 0; k0 < I_; k0 += 32) {
        __syncthreads();

        // ---- stage A tile (already bf16): async global->LDS b128 per lane
#pragma unroll
        for (int r = 0; r < 2; ++r) {
            int t   = tid + (r << 8); // 0..511
            int row = t >> 2;         // 0..127
            int seg = t & 3;          // 8-halfword segment
            const unsigned short* src = A + (size_t)row * I_ + k0 + seg * 8;
            unsigned short* dst = &sA[row * 32 + seg * 8];
#ifdef HAVE_ASYNC_LDS
            __builtin_amdgcn_global_load_async_to_lds_b128(
                (g4p_t)(i4*)src, (l4p_t)(i4*)dst, 0, 0);
#else
            *(u4*)dst = *(const u4*)src;
#endif
            if (k0 + 32 < I_) __builtin_prefetch(src + 32, 0, 0);
        }

        // ---- stage down_proj tile transposed, f32 -> bf16
#pragma unroll
        for (int r = 0; r < 2; ++r) {
            int t  = tid + (r << 8);
            int kp = t & 15;
            int ng = t >> 4;
            const float* dp = D + (size_t)(k0 + kp * 2) * H_ + n0 + ng * 4;
            f4 d0 = ((const f4*)dp)[0];
            f4 d1 = *(const f4*)(dp + H_);
            int base = (ng * 4) * 32 + kp * 2;
            *(unsigned*)&sB[base     ] = pack2(d0.x, d1.x);
            *(unsigned*)&sB[base + 32] = pack2(d0.y, d1.y);
            *(unsigned*)&sB[base + 64] = pack2(d0.z, d1.z);
            *(unsigned*)&sB[base + 96] = pack2(d0.w, d1.w);
            if (k0 + 32 < I_) __builtin_prefetch(dp + (size_t)32 * H_, 0, 0);
        }
#ifdef HAVE_ASYNC_LDS
        wait_asynccnt0(); // async LDS writes visible before the barrier
#endif
        __syncthreads();

        Frag16 a[2];
#pragma unroll
        for (int mi = 0; mi < 2; ++mi) {
            int base = (wm * 32 + mi * 16 + lrow) * 32 + khalf * 8;
            a[mi].u[0] = *(const u4*)&sA[base];
            a[mi].u[1] = *(const u4*)&sA[base + 16];
        }
#pragma unroll
        for (int ni = 0; ni < 4; ++ni) {
            int nb = (wn * 64 + ni * 16 + lrow) * 32 + khalf * 8;
            Frag16 b;
            b.u[0] = *(const u4*)&sB[nb];
            b.u[1] = *(const u4*)&sB[nb + 16];
#pragma unroll
            for (int mi = 0; mi < 2; ++mi) {
                acc[mi][ni] = __builtin_amdgcn_wmma_f32_16x16x32_bf16(
                    false, a[mi].v, false, b.v, (short)0, acc[mi][ni], false, false);
            }
        }
    }

    // ---- epilogue: f32 store
#pragma unroll
    for (int mi = 0; mi < 2; ++mi) {
#pragma unroll
        for (int ni = 0; ni < 4; ++ni) {
            int nn = n0 + wn * 64 + ni * 16 + lrow;
#pragma unroll
            for (int r = 0; r < 8; ++r) {
                int m = tm0 + wm * 32 + mi * 16 + khalf * 8 + r;
                out[((size_t)e * T_ + m) * H_ + nn] = acc[mi][ni][r];
            }
        }
    }
}

extern "C" void kernel_launch(void* const* d_in, const int* in_sizes, int n_in,
                              void* d_out, int out_size, void* d_ws, size_t ws_size,
                              hipStream_t stream) {
    const float* x    = (const float*)d_in[0];
    const float* gate = (const float*)d_in[1];
    const float* up   = (const float*)d_in[2];
    const float* down = (const float*)d_in[3];
    float* out = (float*)d_out;
    unsigned short* act = (unsigned short*)d_ws; // E*T*I bf16 = 22 MiB scratch

    dim3 blk(256);
    dim3 grid1(I_ / 128, T_ / 128, E_); // 11 x 8 x 8
    dim3 grid2(H_ / 128, T_ / 128, E_); // 16 x 8 x 8
    gateup_kernel<<<grid1, blk, 0, stream>>>(x, gate, up, act);
    down_kernel<<<grid2, blk, 0, stream>>>(act, down, out);
}
